// OrphicEmbeddings_7541962572259
// MI455X (gfx1250) — matrix operand order repaired and checked by reading, not compiled, not run on hardware
//
#include <hip/hip_runtime.h>

// OrphicEmbeddings loss for MI455X (gfx1250, wave32).
// Gather-bound workload (~460 MB of embedding-row gathers, W_fwd L2-resident);
// scores computed with v_wmma_f32_16x16x32_f16 (f16 operands, f32 accumulate),
// 16 batch rows per wave, diagonal extraction of the 16x16 WMMA tile.

typedef __attribute__((ext_vector_type(16))) _Float16 v16h;
typedef __attribute__((ext_vector_type(8)))  float    v8f;

#define DIM   256
#define BATCH 16384
#define CTX   20
#define NNEG  5
#define NSLOT (CTX + NNEG)
#define ALPHA 0.5f
#define EPSL  1e-6f

#define WAVES_PER_BLOCK 4
#define THREADS (WAVES_PER_BLOCK * 32)
#define NGROUP  (BATCH / 16)                  // 1024 wave-groups of 16 batch rows
#define NBLOCK  (NGROUP / WAVES_PER_BLOCK)    // 256 blocks

union H8 { int4 i4; _Float16 h[8]; };

__device__ __forceinline__ float fast_sigmoid(float x) {
  return 1.0f / (1.0f + __expf(-x));
}

__global__ __launch_bounds__(THREADS)
void orphic_score_kernel(const int*   __restrict__ tgt,
                         const int*   __restrict__ ctx,
                         const int*   __restrict__ neg,
                         const float* __restrict__ freq,
                         const float* __restrict__ Wf,
                         const float* __restrict__ Wr,
                         const float* __restrict__ Wi,
                         float*       __restrict__ partial) {
  // 16 rows x 256 dims of f16 orphic per wave: 8 KB; 4 waves -> 32 KB LDS.
  __shared__ __align__(16) _Float16 s_orph[WAVES_PER_BLOCK][16 * DIM];
  __shared__ float s_red[WAVES_PER_BLOCK];

  const int wave  = threadIdx.x >> 5;
  const int lane  = threadIdx.x & 31;
  const int group = blockIdx.x * WAVES_PER_BLOCK + wave;
  const int b0    = group * 16;
  const int mrow  = lane & 15;   // matrix row / column index handled by lane
  const int half  = lane >> 4;   // lane-half selects K sub-range per ISA layout

  _Float16* orow = s_orph[wave];

  // ---- Stage 1: build orphic embeddings (16 rows x 256) in LDS as f16 ----
  // One row per iteration; each lane covers 8 consecutive dims (2x float4 loads
  // per table, fully coalesced: 32 lanes x 32B = 1 KB row).
  for (int r = 0; r < 16; ++r) {
    const int   t  = tgt[b0 + r];                        // wave-uniform -> s_load
    const float sc = 1.0f / (1.0f + __logf(freq[t] + EPSL));
    const size_t off = (size_t)t * DIM + (size_t)(lane * 8);
    const float4 f0 = *(const float4*)(Wf + off);
    const float4 f1 = *(const float4*)(Wf + off + 4);
    const float4 r0 = *(const float4*)(Wr + off);
    const float4 r1 = *(const float4*)(Wr + off + 4);
    const float4 i0 = *(const float4*)(Wi + off);
    const float4 i1 = *(const float4*)(Wi + off + 4);
    H8 pk;
    pk.h[0] = (_Float16)(ALPHA * f0.x + (1.0f - ALPHA) * r0.x + i0.x * sc);
    pk.h[1] = (_Float16)(ALPHA * f0.y + (1.0f - ALPHA) * r0.y + i0.y * sc);
    pk.h[2] = (_Float16)(ALPHA * f0.z + (1.0f - ALPHA) * r0.z + i0.z * sc);
    pk.h[3] = (_Float16)(ALPHA * f0.w + (1.0f - ALPHA) * r0.w + i0.w * sc);
    pk.h[4] = (_Float16)(ALPHA * f1.x + (1.0f - ALPHA) * r1.x + i1.x * sc);
    pk.h[5] = (_Float16)(ALPHA * f1.y + (1.0f - ALPHA) * r1.y + i1.y * sc);
    pk.h[6] = (_Float16)(ALPHA * f1.z + (1.0f - ALPHA) * r1.z + i1.z * sc);
    pk.h[7] = (_Float16)(ALPHA * f1.w + (1.0f - ALPHA) * r1.w + i1.w * sc);
    *(int4*)(orow + r * DIM + lane * 8) = pk.i4;         // one ds_store_b128
  }
  __syncthreads();

  // ---- Stage 2: A-matrix fragments (16x32 f16 per chunk) from LDS ----
  // ISA layout: lane holds row M=lane%16; v[0..7] = K = 32c + half*8 + 0..7,
  // v[8..15] = K = 32c + 16 + half*8 + 0..7. Two ds_load_b128 per chunk.
  v16h afrag[8];
#pragma unroll
  for (int c = 0; c < 8; ++c) {
    const _Float16* pa = orow + mrow * DIM + c * 32 + half * 8;
    H8 lo, hi;
    lo.i4 = *(const int4*)pa;
    hi.i4 = *(const int4*)(pa + 16);
#pragma unroll
    for (int i = 0; i < 8; ++i) {
      afrag[c][i]     = lo.h[i];
      afrag[c][8 + i] = hi.h[i];
    }
  }

  // ---- Stage 3: 25 score slots; WMMA over K=256; diagonal = 16 scores ----
  // Diagonal D[n][n] of the f32 16x16 C/D tile lives on lanes 0..7 (elem=lane)
  // and lanes 24..31 (elem=lane-24).
  const bool valid = (lane < 8) || (lane >= 24);
  const int  dsel  = lane & 7;
  float acc_pos = 0.0f, acc_neg = 0.0f;

  for (int j = 0; j < NSLOT; ++j) {
    const int bn    = b0 + mrow;  // batch row mapped to B-matrix column mrow
    const int token = (j < CTX) ? ctx[bn * CTX + j]
                                : neg[bn * NNEG + (j - CTX)];
    const float* wrow = Wf + (size_t)token * DIM;

    v8f acc = {};
#pragma unroll
    for (int c = 0; c < 8; ++c) {
      // B layout (32x16 f16): lane holds column N=lane%16,
      // v[0..7] = K = 32c + half*8 + 0..7, v[8..15] = K = 32c + 16 + half*8 + 0..7.
      const float* p = wrow + c * 32 + half * 8;
      const float4 x0 = *(const float4*)(p);
      const float4 x1 = *(const float4*)(p + 4);
      const float4 x2 = *(const float4*)(p + 16);
      const float4 x3 = *(const float4*)(p + 20);
      v16h bf;
      bf[0]  = (_Float16)x0.x; bf[1]  = (_Float16)x0.y;
      bf[2]  = (_Float16)x0.z; bf[3]  = (_Float16)x0.w;
      bf[4]  = (_Float16)x1.x; bf[5]  = (_Float16)x1.y;
      bf[6]  = (_Float16)x1.z; bf[7]  = (_Float16)x1.w;
      bf[8]  = (_Float16)x2.x; bf[9]  = (_Float16)x2.y;
      bf[10] = (_Float16)x2.z; bf[11] = (_Float16)x2.w;
      bf[12] = (_Float16)x3.x; bf[13] = (_Float16)x3.y;
      bf[14] = (_Float16)x3.z; bf[15] = (_Float16)x3.w;
      acc = __builtin_amdgcn_wmma_f32_16x16x32_f16(
          /*neg_a=*/false, afrag[c], /*neg_b=*/false, bf,
          /*c_mod=*/(short)0, acc, /*reuse_a=*/false, /*reuse_b=*/false);
    }

    // branchless diagonal select (v_cndmask chain, keeps EXEC all-ones)
    float s = acc[0];
#pragma unroll
    for (int i = 1; i < 8; ++i) s = (dsel == i) ? acc[i] : s;

    const float lp = -__logf(fast_sigmoid(s)  + EPSL);
    const float ln = -__logf(fast_sigmoid(-s) + EPSL);
    acc_pos += (valid && (j <  CTX)) ? lp : 0.0f;
    acc_neg += (valid && (j >= CTX)) ? ln : 0.0f;
  }

  // Reference weighting: pos -> mean over B*CTX; neg -> mean over K, sum over B.
  float total = acc_pos * (1.0f / ((float)BATCH * (float)CTX))
              + acc_neg * (1.0f / (float)NNEG);

  // Deterministic fixed-order wave reduction (wave32).
#pragma unroll
  for (int m = 16; m >= 1; m >>= 1) total += __shfl_xor(total, m, 32);
  if (lane == 0) s_red[wave] = total;
  __syncthreads();
  if (threadIdx.x == 0) {
    float blk = 0.0f;
    for (int w = 0; w < WAVES_PER_BLOCK; ++w) blk += s_red[w];
    partial[blockIdx.x] = blk;
  }
}

__global__ void orphic_reduce_kernel(const float* __restrict__ partial,
                                     float* __restrict__ out) {
  if (threadIdx.x == 0 && blockIdx.x == 0) {
    float s = 0.0f;
    for (int i = 0; i < NBLOCK; ++i) s += partial[i];  // fixed order, deterministic
    out[0] = s;
  }
}

extern "C" void kernel_launch(void* const* d_in, const int* in_sizes, int n_in,
                              void* d_out, int out_size, void* d_ws, size_t ws_size,
                              hipStream_t stream) {
  const int*   tgt  = (const int*)  d_in[0];  // target_tokens   [B]
  const int*   ctx  = (const int*)  d_in[1];  // context_tokens  [B, 20]
  const int*   neg  = (const int*)  d_in[2];  // negative_samples[B, 5]
  const float* freq = (const float*)d_in[3];  // token_frequencies [V]
  const float* Wf   = (const float*)d_in[4];  // W_fwd [V, 256]
  const float* Wr   = (const float*)d_in[5];  // W_rev [V, 256]
  const float* Wi   = (const float*)d_in[6];  // W_iso [V, 256]
  float*       out  = (float*)d_out;          // scalar loss
  float*       partial = (float*)d_ws;        // NBLOCK floats of scratch

  orphic_score_kernel<<<NBLOCK, THREADS, 0, stream>>>(tgt, ctx, neg, freq,
                                                      Wf, Wr, Wi, partial);
  orphic_reduce_kernel<<<1, 32, 0, stream>>>(partial, out);
}
